// HydraBlock_67980742361430
// MI455X (gfx1250) — matrix operand order repaired and checked by reading, not compiled, not run on hardware
//
#include <hip/hip_runtime.h>
#include <hip/hip_bf16.h>
#include <math.h>

// ---------------- problem constants ----------------
#define D_MODEL   1024
#define D_STATE   64
#define D_CONV    7
#define HEADDIM   64
#define NHEADS    32
#define D_INNER   2048
#define D_IN_PROJ 4416               // 2*2048 + 2*(2*64) + 2*32
#define CONV_DIM  2304               // 2048 + 256
#define BATCH     2
#define SEQLEN    2048
#define ROWS      (BATCH * SEQLEN)   // 4096
#define EPS       1e-5f

// ---------------- vector types for WMMA ----------------
typedef __attribute__((ext_vector_type(16))) __bf16        v16bf;
typedef __attribute__((ext_vector_type(8)))  float         v8f;
typedef __attribute__((ext_vector_type(4)))  unsigned int  u32x4;

// A fragment (16x32 bf16, MxK): lane = M row; low 8 elems at k0, high 8 at k0+16
__device__ __forceinline__ v16bf load_a_frag(const __bf16* p) {
  v16bf r;
  ((u32x4*)&r)[0] = *(const u32x4*)(p);
  ((u32x4*)&r)[1] = *(const u32x4*)(p + 16);
  return r;
}
// B fragment (32x16 bf16, KxN): lane = N col; low 8 elems at k0, high 8 at k0+16
__device__ __forceinline__ v16bf load_b_frag(const __bf16* p) {
  v16bf r;
  ((u32x4*)&r)[0] = *(const u32x4*)(p);
  ((u32x4*)&r)[1] = *(const u32x4*)(p + 16);
  return r;
}

// gfx1250 async global->LDS copy (16B per lane), tracked by ASYNCcnt.
// LDS offset for VDST = flat addr[31:0].
__device__ __forceinline__ void async_copy_b128(const void* g, void* l) {
  unsigned lds_off = (unsigned)(unsigned long long)l;
  asm volatile("global_load_async_to_lds_b128 %0, %1, off"
               :: "v"(lds_off), "v"(g) : "memory");
}
__device__ __forceinline__ void wait_async0() {
  asm volatile("s_wait_asynccnt 0x0" ::: "memory");
}

// ---------------- fp32 -> bf16 cast ----------------
__global__ __launch_bounds__(256) void cast_bf16_kernel(const float* __restrict__ in,
                                                        __bf16* __restrict__ out, int n) {
  int i = blockIdx.x * 256 + threadIdx.x;
  if (i < n) out[i] = (__bf16)in[i];
}

// ---------------- WMMA GEMM: C[M,N] = A[M,K] * B[N,K]^T ----------------
// block = 256 threads (8 waves along M). Wave tile: 32(M) x 64(N), 8 accumulators.
// B tile (64 x 32 bf16) staged into LDS by async copies, double-buffered.
// grid = (N/64, M/256). M%256==0, N%64==0, K%32==0.
#define BROW_STRIDE 80   // bytes; padded so ds_load_b128 across 16 rows is conflict-free
__global__ __launch_bounds__(256) void gemm_bf16_nt(float* __restrict__ C,
                                                    const __bf16* __restrict__ A,
                                                    const __bf16* __restrict__ B,
                                                    int M, int N, int K) {
  __shared__ __align__(16) char lds_b[2][64 * BROW_STRIDE];   // 2 x 5 KB

  const int tid  = threadIdx.x;
  const int lane = tid & 31;
  const int wave = tid >> 5;
  const int hi   = lane >> 4;
  const int r    = lane & 15;

  const int mBase = blockIdx.y * 256 + wave * 32;
  const int nBase = blockIdx.x * 64;

  // async staging assignment: 256 threads cover 64 rows x 4 x 16B
  const int srow = tid >> 2, sq = tid & 3;
  const __bf16* gB = B + (size_t)(nBase + srow) * K + sq * 8;
  char* lB0 = &lds_b[0][srow * BROW_STRIDE + sq * 16];
  char* lB1 = &lds_b[1][srow * BROW_STRIDE + sq * 16];

  const __bf16* Arow0 = A + (size_t)(mBase + r) * K;       // lane's M row, tile 0
  const __bf16* Arow1 = Arow0 + (size_t)16 * K;            // tile 1

  v8f acc[8];
#pragma unroll
  for (int i = 0; i < 8; ++i) acc[i] = (v8f){};

  const int nK = K >> 5;
  async_copy_b128(gB, lB0);                                // stage 0 -> buf 0

  for (int i = 0; i < nK; ++i) {
    wait_async0();              // stage i resident in buf[i&1]
    __syncthreads();            // visible to all waves; prev compute done
    if (i + 1 < nK)             // overlap stage i+1 copy with compute
      async_copy_b128(gB + (size_t)(i + 1) * 32, (i & 1) ? lB0 : lB1);

    const char* lbuf = lds_b[i & 1];
    const int ka = i * 32 + hi * 8;
    __builtin_prefetch(Arow0 + ka + 256, 0, 0);            // global_prefetch_b8
    __builtin_prefetch(Arow1 + ka + 256, 0, 0);
    v16bf a0 = load_a_frag(Arow0 + ka);
    v16bf a1 = load_a_frag(Arow1 + ka);

#pragma unroll
    for (int nt = 0; nt < 4; ++nt) {
      v16bf bf = load_b_frag((const __bf16*)(lbuf + (nt * 16 + r) * BROW_STRIDE + hi * 16));
      acc[nt]     = __builtin_amdgcn_wmma_f32_16x16x32_bf16(false, a0, false, bf, (short)0, acc[nt],     false, false);
      acc[4 + nt] = __builtin_amdgcn_wmma_f32_16x16x32_bf16(false, a1, false, bf, (short)0, acc[4 + nt], false, false);
    }
  }

  // D layout: lane -> N = r, VGPR v -> M = v + 8*hi
#pragma unroll
  for (int mt = 0; mt < 2; ++mt) {
    float* Crow = C + (size_t)(mBase + mt * 16 + 8 * hi) * N + nBase + r;
#pragma unroll
    for (int v = 0; v < 8; ++v)
#pragma unroll
      for (int nt = 0; nt < 4; ++nt)
        Crow[(size_t)v * N + nt * 16] = acc[mt * 4 + nt][v];
  }
}

// ---------------- depthwise conv7 + SiLU on xBC slice of zxbcdt ----------------
__global__ __launch_bounds__(256) void conv_silu_kernel(const float* __restrict__ zxbcdt,
                                                        const float* __restrict__ conv_w,
                                                        const float* __restrict__ conv_b,
                                                        float* __restrict__ xBC_act) {
  int idx = blockIdx.x * 256 + threadIdx.x;          // over ROWS*CONV_DIM (exact)
  int c  = idx % CONV_DIM;
  int bl = idx / CONV_DIM;
  int l  = bl % SEQLEN;
  int b  = bl / SEQLEN;

  float acc = conv_b[c];
#pragma unroll
  for (int j = 0; j < D_CONV; ++j) {
    int ll = l + j - (D_CONV / 2);
    if (ll >= 0 && ll < SEQLEN) {
      acc += zxbcdt[((size_t)(b * SEQLEN + ll)) * D_IN_PROJ + D_INNER + c] * conv_w[c * D_CONV + j];
    }
  }
  xBC_act[(size_t)bl * CONV_DIM + c] = acc / (1.f + __expf(-acc));   // SiLU
}

// ---------------- diag_heads[bl,h] = dot(x_og[bl,:], fc_D_w[h,:]) + Dp[h] -------
__global__ __launch_bounds__(256) void diag_kernel(const float* __restrict__ xBC_act,
                                                   const float* __restrict__ fc_D_w,
                                                   const float* __restrict__ Dp,
                                                   float* __restrict__ diag) {
  int idx = blockIdx.x * 256 + threadIdx.x;          // ROWS*NHEADS (exact)
  int h  = idx & (NHEADS - 1);
  int bl = idx >> 5;
  const float4* xr = (const float4*)(xBC_act + (size_t)bl * CONV_DIM);
  const float4* wr = (const float4*)(fc_D_w + (size_t)h * D_INNER);
  float acc = 0.f;
  for (int k = 0; k < D_INNER / 4; ++k) {
    float4 x = xr[k], w = wr[k];
    acc += x.x * w.x + x.y * w.y + x.z * w.z + x.w * w.w;
  }
  diag[idx] = acc + Dp[h];
}

__device__ __forceinline__ float softplusf(float x) {
  return (x > 20.f) ? x : log1pf(__expf(x));
}

// ---------------- sequential selective scan ----------------
// grid = 128 (sb in 0..3, h in 0..31); block = 32 (one wave; lane owns p, p+32).
// Barrier-free: intra-wave LDS broadcast (same-wave DS ops are in-order).
// Next timestep's globals prefetched into registers before the compute loop.
// y_scan layout: [sb][L][D_INNER], shifted by one (row 0 zeroed, y_t -> row t+1)
__global__ __launch_bounds__(32) void scan_kernel(const float* __restrict__ xBC_act,
                                                  const float* __restrict__ zxbcdt,
                                                  const float* __restrict__ dt_bias,
                                                  const float* __restrict__ A_log,
                                                  float* __restrict__ y_scan) {
  const int h    = blockIdx.x & (NHEADS - 1);
  const int sb   = blockIdx.x >> 5;
  const int b    = sb & 1;
  const int dir  = sb >> 1;
  const int lane = threadIdx.x;

  __shared__ __align__(8) float BCs[2 * D_STATE];    // interleaved {B[s], C[s]}

  float st0[D_STATE], st1[D_STATE];
#pragma unroll
  for (int s = 0; s < D_STATE; ++s) { st0[s] = 0.f; st1[s] = 0.f; }

  const float A    = -__expf(A_log[h]);
  const float bias = dt_bias[h];
  const int   bcoff = D_INNER + (dir ? 128 : 0);     // B at +0/+128, C at +64/+192
  const int   dtcol = D_INNER + CONV_DIM + (dir ? NHEADS : 0) + h;
  const size_t ybase = (size_t)sb * SEQLEN * D_INNER;

  // shifted row 0 = zeros
  y_scan[ybase + h * HEADDIM + lane] = 0.f;
  y_scan[ybase + h * HEADDIM + lane + 32] = 0.f;

  // preload t = 0
  int tt = dir ? (SEQLEN - 1) : 0;
  size_t rowx = (size_t)(b * SEQLEN + tt) * CONV_DIM;
  size_t rowz = (size_t)(b * SEQLEN + tt) * D_IN_PROJ;
  float nb0 = xBC_act[rowx + bcoff + lane];
  float nb1 = xBC_act[rowx + bcoff + lane + 32];
  float nc0 = xBC_act[rowx + bcoff + D_STATE + lane];
  float nc1 = xBC_act[rowx + bcoff + D_STATE + lane + 32];
  float nx0 = xBC_act[rowx + (size_t)h * HEADDIM + lane];
  float nx1 = xBC_act[rowx + (size_t)h * HEADDIM + lane + 32];
  float ndt = zxbcdt[rowz + dtcol];

  for (int t = 0; t < SEQLEN; ++t) {
    // broadcast current B/C through LDS (single wave: no barrier needed)
    BCs[2 * lane]              = nb0;
    BCs[2 * lane + 1]          = nc0;
    BCs[2 * (lane + 32)]       = nb1;
    BCs[2 * (lane + 32) + 1]   = nc1;
    const float x0 = nx0, x1 = nx1, dtr = ndt;
    __builtin_amdgcn_wave_barrier();                 // pin schedule: stores before reads

    if (t + 1 < SEQLEN) {                            // prefetch t+1 (hides global latency)
      const int tn = dir ? (SEQLEN - 2 - t) : (t + 1);
      rowx = (size_t)(b * SEQLEN + tn) * CONV_DIM;
      rowz = (size_t)(b * SEQLEN + tn) * D_IN_PROJ;
      nb0 = xBC_act[rowx + bcoff + lane];
      nb1 = xBC_act[rowx + bcoff + lane + 32];
      nc0 = xBC_act[rowx + bcoff + D_STATE + lane];
      nc1 = xBC_act[rowx + bcoff + D_STATE + lane + 32];
      nx0 = xBC_act[rowx + (size_t)h * HEADDIM + lane];
      nx1 = xBC_act[rowx + (size_t)h * HEADDIM + lane + 32];
      ndt = zxbcdt[rowz + dtcol];
    }

    const float dt = softplusf(dtr + bias);
    const float ex = __expf(A * dt);
    float acc0 = 0.f, acc1 = 0.f;
#pragma unroll
    for (int s = 0; s < D_STATE; ++s) {
      float2 bc = *(const float2*)&BCs[2 * s];       // ds_load_b64 broadcast
      st0[s] = st0[s] * ex + x0 * bc.x;  acc0 += st0[s] * bc.y;
      st1[s] = st1[s] * ex + x1 * bc.x;  acc1 += st1[s] * bc.y;
    }
    if (t + 1 < SEQLEN) {
      y_scan[ybase + (size_t)(t + 1) * D_INNER + h * HEADDIM + lane]      = acc0;
      y_scan[ybase + (size_t)(t + 1) * D_INNER + h * HEADDIM + lane + 32] = acc1;
    }
    __builtin_amdgcn_wave_barrier();                 // reads of BCs before next stores
  }
}

// ---------------- combine + RMS norm + SiLU(z) gate, emit bf16 ----------------
__global__ __launch_bounds__(256) void combine_norm_kernel(const float* __restrict__ y_scan,
                                                           const float* __restrict__ xBC_act,
                                                           const float* __restrict__ zxbcdt,
                                                           const float* __restrict__ diag,
                                                           const float* __restrict__ norm_w,
                                                           __bf16* __restrict__ y_bf16) {
  const int bl = blockIdx.x;
  const int b  = bl / SEQLEN;
  const int l  = bl % SEQLEN;

  const size_t fwrow = (size_t)(b * SEQLEN + l) * D_INNER;
  const size_t bwrow = (size_t)((2 + b) * SEQLEN + (SEQLEN - 1 - l)) * D_INNER;
  const size_t xrow  = (size_t)bl * CONV_DIM;
  const size_t zrow  = (size_t)bl * D_IN_PROJ;
  const float* dg = diag + (size_t)bl * NHEADS;

  __shared__ float red[256];
  float vals[8];
  float ss = 0.f;
#pragma unroll
  for (int i = 0; i < 8; ++i) {
    int d = threadIdx.x + i * 256;
    float x = xBC_act[xrow + d];
    float v = y_scan[fwrow + d] + y_scan[bwrow + d] + x * dg[d >> 6];
    vals[i] = v;
    ss += v * v;
  }
  red[threadIdx.x] = ss;
  __syncthreads();
  for (int off = 128; off > 0; off >>= 1) {
    if (threadIdx.x < off) red[threadIdx.x] += red[threadIdx.x + off];
    __syncthreads();
  }
  const float rinv = rsqrtf(red[0] / (float)D_INNER + EPS);
#pragma unroll
  for (int i = 0; i < 8; ++i) {
    int d = threadIdx.x + i * 256;
    float z  = zxbcdt[zrow + d];
    float sz = z / (1.f + __expf(-z));
    y_bf16[(size_t)bl * D_INNER + d] = (__bf16)(norm_w[d] * vals[i] * rinv * sz);
  }
}

// ---------------- launch ----------------
extern "C" void kernel_launch(void* const* d_in, const int* in_sizes, int n_in,
                              void* d_out, int out_size, void* d_ws, size_t ws_size,
                              hipStream_t stream) {
  const float* u       = (const float*)d_in[0];
  const float* W_in    = (const float*)d_in[1];
  const float* conv_w  = (const float*)d_in[2];
  const float* conv_b  = (const float*)d_in[3];
  const float* dt_bias = (const float*)d_in[4];
  const float* A_log   = (const float*)d_in[5];
  const float* Dp      = (const float*)d_in[6];
  const float* fc_D_w  = (const float*)d_in[7];
  const float* norm_w  = (const float*)d_in[8];
  const float* W_out   = (const float*)d_in[9];
  float* out = (float*)d_out;

  // workspace carve-up (256B aligned)
  char* ws = (char*)d_ws;
  __bf16* u_bf    = (__bf16*)(ws);                                   //  8.39 MB
  __bf16* Win_bf  = (__bf16*)(ws + 8388608);                         //  9.04 MB
  __bf16* Wout_bf = (__bf16*)(ws + 17432576);                        //  4.19 MB
  float*  zxbcdt  = (float*) (ws + 21626880);                        // 72.35 MB
  float*  xBC_act = (float*) (ws + 93978624);                        // 37.75 MB
  float*  diag    = (float*) (ws + 131727360);                       //  0.52 MB
  float*  y_scan  = (float*) (ws + 132251648);                       // 67.11 MB
  __bf16* y_bf    = (__bf16*)(ws + 199360512);                       // 16.78 MB

  // 1) fp32 -> bf16 casts
  {
    int n;
    n = ROWS * D_MODEL;
    cast_bf16_kernel<<<(n + 255) / 256, 256, 0, stream>>>(u, u_bf, n);
    n = D_IN_PROJ * D_MODEL;
    cast_bf16_kernel<<<(n + 255) / 256, 256, 0, stream>>>(W_in, Win_bf, n);
    n = D_MODEL * D_INNER;
    cast_bf16_kernel<<<(n + 255) / 256, 256, 0, stream>>>(W_out, Wout_bf, n);
  }

  // 2) in-projection GEMM: zxbcdt[4096,4416] = u[4096,1024] * W_in^T
  gemm_bf16_nt<<<dim3(D_IN_PROJ / 64, ROWS / 256), 256, 0, stream>>>(
      zxbcdt, u_bf, Win_bf, ROWS, D_IN_PROJ, D_MODEL);

  // 3) depthwise conv7 + SiLU
  conv_silu_kernel<<<(ROWS * CONV_DIM) / 256, 256, 0, stream>>>(zxbcdt, conv_w, conv_b, xBC_act);

  // 4) diag heads GEMV
  diag_kernel<<<(ROWS * NHEADS) / 256, 256, 0, stream>>>(xBC_act, fc_D_w, Dp, diag);

  // 5) bidirectional selective scan (one wave per (sb, head))
  scan_kernel<<<4 * NHEADS, 32, 0, stream>>>(xBC_act, zxbcdt, dt_bias, A_log, y_scan);

  // 6) combine + RMS norm + gate -> bf16
  combine_norm_kernel<<<ROWS, 256, 0, stream>>>(y_scan, xBC_act, zxbcdt, diag, norm_w, y_bf);

  // 7) out-projection GEMM: out[4096,1024] = y[4096,2048] * W_out^T
  gemm_bf16_nt<<<dim3(D_MODEL / 64, ROWS / 256), 256, 0, stream>>>(
      out, y_bf, Wout_bf, ROWS, D_MODEL, D_INNER);
}
